// Up_69148973466202
// MI455X (gfx1250) — compile-verified, matrix operand order checked by reading, not run on hardware
//
#include <hip/hip_runtime.h>
#include <hip/hip_bf16.h>
#include <math.h>

// ---------------------------------------------------------------------------
// Types for CDNA5 WMMA (wave32).
// ---------------------------------------------------------------------------
typedef __bf16 bf16;
typedef bf16  bf16x8 __attribute__((ext_vector_type(8)));
typedef bf16  v16bf  __attribute__((ext_vector_type(16)));
typedef float v8f    __attribute__((ext_vector_type(8)));

// Problem constants
#define D32   32
#define DP    36            // padded dim (pad 2 each side)
#define CIN   80
#define CPAD  96            // channels padded to 3x32 K-chunks
#define COUT  96            // CSO(48) + 3*C1(48)
#define NTAP  125
#define NKC   3             // 96/32 K chunks
#define NNB   6             // 96/16 N blocks
#define NVOX  (D32*D32*D32) // 32768
#define NFINE (64*64*64)

#define PANEL_ELEMS (NKC*NNB*32*16)       // 9216 bf16 per tap
#define PANEL_BYTES (PANEL_ELEMS*2)       // 18432 B
#define LOADS_PER_THREAD (PANEL_BYTES/16/128)  // 9 x b128 per thread (128 thr)

// packed-B: [tap][kc][nb][lane][16]  (bf16)
#define PB_ELEMS ((size_t)NTAP*PANEL_ELEMS)
#define PB_BYTES (PB_ELEMS*2)
#define XPAD_ELEMS ((size_t)DP*DP*DP*CPAD)
#define XPAD_BYTES (XPAD_ELEMS*2)

// ---------------------------------------------------------------------------
// 1) Build packed WMMA B-fragments of the equivariant conv kernel.
//    Fragment layout (16-bit A/B, wave32): lane -> n=lane&15, half=lane>>4,
//    element j -> k = 16*(j>>3) + 8*half + (j&7).
// ---------------------------------------------------------------------------
__global__ void build_pb_kernel(const float* __restrict__ ss,
                                const float* __restrict__ sv,
                                const float* __restrict__ vs,
                                const float* __restrict__ vv0,
                                const float* __restrict__ vv1,
                                bf16* __restrict__ pb) {
  int t = blockIdx.x * blockDim.x + threadIdx.x;
  if (t >= NTAP * NKC * NNB * 32) return;
  const int lane = t & 31;
  int rest = t >> 5;
  const int nb  = rest % NNB;  rest /= NNB;
  const int kc  = rest % NKC;
  const int tap = rest / NKC;

  const int kd = tap / 25, kh = (tap / 5) % 5, kw = tap % 5;
  const float c[3] = {(float)(kd - 2), (float)(kh - 2), (float)(kw - 2)};
  const float norm = sqrtf(c[0]*c[0] + c[1]*c[1] + c[2]*c[2]);
  const float mask = (norm <= 2.5f) ? 1.f : 0.f;
  float R[3];
  #pragma unroll
  for (int r = 0; r < 3; ++r) {
    float dd = (norm - 1.25f * (float)r) * (1.f / 1.25f);
    R[r] = expf(-dd * dd) * mask;
  }
  float Y[3];
  #pragma unroll
  for (int m = 0; m < 3; ++m)
    Y[m] = (norm == 0.f) ? 0.f : 1.7320508075688772f * c[m] / norm;

  const int half = lane >> 4;
  const int n    = lane & 15;
  const int cout = nb * 16 + n;
  bf16* out = pb + (size_t)t * 16;

  #pragma unroll
  for (int j = 0; j < 16; ++j) {
    const int k   = 16 * (j >> 3) + 8 * half + (j & 7);
    const int cin = kc * 32 + k;
    float v = 0.f;
    if (cin < 32) {
      if (cout < 48) {
        float s = 0.f;
        #pragma unroll
        for (int r = 0; r < 3; ++r) s += R[r] * ss[(r * 32 + cin) * 48 + cout];
        v = s;
      } else {
        const int oc = (cout - 48) / 3, m = (cout - 48) % 3;
        float s = 0.f;
        #pragma unroll
        for (int r = 0; r < 3; ++r) s += R[r] * sv[(r * 32 + cin) * 16 + oc];
        v = s * Y[m];
      }
    } else if (cin < 80) {
      const int ic = (cin - 32) / 3, a = (cin - 32) % 3;
      if (cout < 48) {
        float s = 0.f;
        #pragma unroll
        for (int r = 0; r < 3; ++r) s += R[r] * vs[(r * 16 + ic) * 48 + cout];
        v = s * Y[a] * 0.5773502691896258f;  // /sqrt(3)
      } else {
        const int oc = (cout - 48) / 3, b = (cout - 48) % 3;
        float k0 = 0.f, k1 = 0.f;
        #pragma unroll
        for (int r = 0; r < 3; ++r) {
          k0 += R[r] * vv0[(r * 16 + ic) * 16 + oc];
          k1 += R[r] * vv1[(r * 16 + ic) * 16 + oc];
        }
        float e1 = 0.f;
        if (a != b) {
          const int m = 3 - a - b;                            // nonzero eps idx
          const float sgn = (m == (a + 1) % 3) ? 1.f : -1.f;  // eps_{a,m,b}
          e1 = sgn * Y[m] * 0.7071067811865476f;              // /sqrt(2)
        }
        v = (a == b ? k0 : 0.f) + k1 * e1;
      }
    }
    out[j] = (bf16)v;
  }
}

// ---------------------------------------------------------------------------
// 2) Pad + convert input: x[32^3][80] f32 -> xpad[36^3][96] bf16
// ---------------------------------------------------------------------------
__global__ void pad_input_kernel(const float* __restrict__ x,
                                 bf16* __restrict__ xpad) {
  int p = blockIdx.x * blockDim.x + threadIdx.x;
  if (p >= DP * DP * DP) return;
  const int pw = p % DP, ph = (p / DP) % DP, pd = p / (DP * DP);
  bf16* o = xpad + (size_t)p * CPAD;
  const bool inb = (pd >= 2 && pd < 34 && ph >= 2 && ph < 34 && pw >= 2 && pw < 34);
  if (inb) {
    const float* xi = x + (size_t)(((pd - 2) * D32 + (ph - 2)) * D32 + (pw - 2)) * CIN;
    for (int cc = 0; cc < CIN; ++cc) o[cc] = (bf16)xi[cc];
    for (int cc = CIN; cc < CPAD; ++cc) o[cc] = (bf16)0.f;
  } else {
    for (int cc = 0; cc < CPAD; ++cc) o[cc] = (bf16)0.f;
  }
}

// ---------------------------------------------------------------------------
// 3) Implicit-GEMM conv via v_wmma_f32_16x16x32_bf16.
//    One wave per 16-voxel M-tile computing all 96 output channels (6 acc
//    tiles, A reused 6x). The per-tap B panel (18KB) is staged into LDS with
//    double-buffered global_load_async_to_lds_b128 shared by the 4 waves of
//    the block.
// ---------------------------------------------------------------------------
__device__ __forceinline__ void issue_panel_async(const bf16* __restrict__ pb,
                                                  int tap, bf16* lds_dst,
                                                  int tid) {
  const bf16* src = pb + (size_t)tap * PANEL_ELEMS;   // wave-uniform SGPR base
  // low 32 bits of a flat LDS pointer == wave-relative LDS byte offset
  unsigned ldsbase = (unsigned)(unsigned long long)(const void*)lds_dst;
  #pragma unroll
  for (int i = 0; i < LOADS_PER_THREAD; ++i) {
    unsigned off = (unsigned)(tid * 16 + i * 2048);   // 128 thr * 16B stride
    asm volatile("global_load_async_to_lds_b128 %0, %1, %2"
                 :: "v"(ldsbase + off), "v"(off), "s"(src)
                 : "memory");
  }
}

__global__ __launch_bounds__(128)
void conv_wmma_kernel(const bf16* __restrict__ xpad,
                      const bf16* __restrict__ pb,
                      float* __restrict__ y) {
  __shared__ __align__(16) bf16 panel[2][PANEL_ELEMS];

  const int lane = threadIdx.x;                       // 0..31
  const int wy   = threadIdx.y;                       // 0..3
  const int tid  = wy * 32 + lane;                    // 0..127
  const int mtile = blockIdx.x * 4 + wy;              // 0..2047
  const int vbase = mtile << 4;
  const int d  = vbase >> 10;
  const int h  = (vbase >> 5) & 31;
  const int w0 = vbase & 31;                          // 0 or 16
  const int half = lane >> 4;
  const int mr   = lane & 15;

  v8f acc[NNB];
  #pragma unroll
  for (int nb = 0; nb < NNB; ++nb) acc[nb] = (v8f){};

  issue_panel_async(pb, 0, &panel[0][0], tid);

  for (int tap = 0; tap < NTAP; ++tap) {
    const int buf = tap & 1;
    if (tap + 1 < NTAP) {
      issue_panel_async(pb, tap + 1, &panel[buf ^ 1][0], tid);
      asm volatile("s_wait_asynccnt 0x9" ::: "memory");  // current tap done
    } else {
      asm volatile("s_wait_asynccnt 0x0" ::: "memory");
    }
    __syncthreads();

    const int kd = tap / 25;
    const int rr = tap - kd * 25;
    const int kh = rr / 5;
    const int kw = rr - kh * 5;
    const bf16* ap = xpad +
        ((size_t)((d + kd) * DP + (h + kh)) * DP + (w0 + mr + kw)) * CPAD + half * 8;
    const bf16* lp = &panel[buf][0];

    #pragma unroll
    for (int kc = 0; kc < NKC; ++kc) {
      bf16x8 alo = *(const bf16x8*)(ap + kc * 32);
      bf16x8 ahi = *(const bf16x8*)(ap + kc * 32 + 16);
      v16bf a = __builtin_shufflevector(alo, ahi,
                  0,1,2,3,4,5,6,7,8,9,10,11,12,13,14,15);
      #pragma unroll
      for (int nb = 0; nb < NNB; ++nb) {
        const bf16* bp = lp + kc * (NNB * 32 * 16) + nb * 512 + lane * 16;
        bf16x8 blo = *(const bf16x8*)bp;
        bf16x8 bhi = *(const bf16x8*)(bp + 8);
        v16bf b = __builtin_shufflevector(blo, bhi,
                    0,1,2,3,4,5,6,7,8,9,10,11,12,13,14,15);
        acc[nb] = __builtin_amdgcn_wmma_f32_16x16x32_bf16(
            false, a, false, b, (short)0, acc[nb], false, false);
      }
    }
    __syncthreads();   // done reading panel[buf] before it is refilled
  }

  // C/D layout: vgpr r -> row = r + 8*half, col = lane&15
  float* yout = y + (size_t)(vbase + half * 8) * COUT + mr;
  #pragma unroll
  for (int nb = 0; nb < NNB; ++nb)
    #pragma unroll
    for (int r = 0; r < 8; ++r)
      yout[(size_t)r * COUT + nb * 16] = acc[nb][r];
}

// ---------------------------------------------------------------------------
// 4) Gate y[96] -> padded bf16 input for conv2 (80 real ch, 16 zero pad).
// ---------------------------------------------------------------------------
__global__ void gate_pad_kernel(const float* __restrict__ y,
                                bf16* __restrict__ xpad) {
  int p = blockIdx.x * blockDim.x + threadIdx.x;
  if (p >= DP * DP * DP) return;
  const int pw = p % DP, ph = (p / DP) % DP, pd = p / (DP * DP);
  bf16* o = xpad + (size_t)p * CPAD;
  const bool inb = (pd >= 2 && pd < 34 && ph >= 2 && ph < 34 && pw >= 2 && pw < 34);
  if (!inb) {
    for (int cc = 0; cc < CPAD; ++cc) o[cc] = (bf16)0.f;
    return;
  }
  const int v = ((pd - 2) * D32 + (ph - 2)) * D32 + (pw - 2);
  const float* yi = y + (size_t)v * COUT;
  for (int cc = 0; cc < 32; ++cc) o[cc] = (bf16)fmaxf(yi[cc], 0.f);
  for (int cc = 0; cc < 16; ++cc) {
    const float g = 1.f / (1.f + expf(-yi[32 + cc]));
    #pragma unroll
    for (int m = 0; m < 3; ++m)
      o[32 + cc * 3 + m] = (bf16)(yi[48 + cc * 3 + m] * g);
  }
  for (int cc = CIN; cc < CPAD; ++cc) o[cc] = (bf16)0.f;
}

// 5) Gate y[96] -> compact f32 [32^3][80] for the head.
__global__ void gate_out_kernel(const float* __restrict__ y,
                                float* __restrict__ g2) {
  int v = blockIdx.x * blockDim.x + threadIdx.x;
  if (v >= NVOX) return;
  const float* yi = y + (size_t)v * COUT;
  float* o = g2 + (size_t)v * CIN;
  for (int cc = 0; cc < 32; ++cc) o[cc] = fmaxf(yi[cc], 0.f);
  for (int cc = 0; cc < 16; ++cc) {
    const float g = 1.f / (1.f + expf(-yi[32 + cc]));
    #pragma unroll
    for (int m = 0; m < 3; ++m)
      o[32 + cc * 3 + m] = yi[48 + cc * 3 + m] * g;
  }
}

// ---------------------------------------------------------------------------
// 6) Upsample x2 + position tensor products + linear head -> out[64^3][80].
// ---------------------------------------------------------------------------
__global__ void final_kernel(const float* __restrict__ g2,
                             const float* __restrict__ wpos,
                             const float* __restrict__ wls,   // [16][32]
                             const float* __restrict__ wlv,   // [48][16]
                             float* __restrict__ out) {
  int f = blockIdx.x * blockDim.x + threadIdx.x;
  if (f >= NFINE) return;
  const int fw = f & 63, fh = (f >> 6) & 63, fd = f >> 12;
  const int cd = fd >> 1, ch = fh >> 1, cw = fw >> 1;
  const float wp = wpos[0];
  const float p0 = wp * (float)(fd - (cd + 16));
  const float p1 = wp * (float)(fh - (ch + 16));
  const float p2 = wp * (float)(fw - (cw + 16));

  const float* xi = g2 + (size_t)((cd * D32 + ch) * D32 + cw) * CIN;
  float S[32];
  #pragma unroll 8
  for (int i = 0; i < 32; ++i) S[i] = xi[i];
  float V[16][3];
  #pragma unroll 4
  for (int cc = 0; cc < 16; ++cc)
    for (int m = 0; m < 3; ++m) V[cc][m] = xi[32 + cc * 3 + m];

  // t_s and S_out
  float ts[16];
  #pragma unroll 4
  for (int cc = 0; cc < 16; ++cc)
    ts[cc] = (V[cc][0] * p0 + V[cc][1] * p1 + V[cc][2] * p2) * 0.5773502691896258f;
  float* of = out + (size_t)f * CIN;
  for (int o = 0; o < 32; ++o) {
    float s = 0.f;
    #pragma unroll 4
    for (int cc = 0; cc < 16; ++cc) s += ts[cc] * wls[cc * 32 + o];
    of[o] = s * 0.25f;                           // 1/sqrt(C1)
  }

  // V_out: tv rows = [ S[c]*p (32), cross(V[c],p)/sqrt2 (16) ], wlv [48][16]
  float a0[16], a1[16], a2[16];
  #pragma unroll 4
  for (int o = 0; o < 16; ++o) { a0[o] = 0.f; a1[o] = 0.f; a2[o] = 0.f; }
  for (int cc = 0; cc < 32; ++cc) {
    const float t0 = S[cc] * p0, t1 = S[cc] * p1, t2 = S[cc] * p2;
    #pragma unroll 4
    for (int o = 0; o < 16; ++o) {
      const float w = wlv[cc * 16 + o];
      a0[o] += t0 * w; a1[o] += t1 * w; a2[o] += t2 * w;
    }
  }
  for (int cc = 0; cc < 16; ++cc) {
    const float cr0 = (V[cc][1] * p2 - V[cc][2] * p1) * 0.7071067811865476f;
    const float cr1 = (V[cc][2] * p0 - V[cc][0] * p2) * 0.7071067811865476f;
    const float cr2 = (V[cc][0] * p1 - V[cc][1] * p0) * 0.7071067811865476f;
    #pragma unroll 4
    for (int o = 0; o < 16; ++o) {
      const float w = wlv[(32 + cc) * 16 + o];
      a0[o] += cr0 * w; a1[o] += cr1 * w; a2[o] += cr2 * w;
    }
  }
  const float inv48 = 0.14433756729740643f;      // 1/sqrt(C0+C1)
  #pragma unroll 4
  for (int o = 0; o < 16; ++o) {
    of[32 + o * 3 + 0] = a0[o] * inv48;
    of[32 + o * 3 + 1] = a1[o] * inv48;
    of[32 + o * 3 + 2] = a2[o] * inv48;
  }
}

// ---------------------------------------------------------------------------
// Host launcher
// ---------------------------------------------------------------------------
extern "C" void kernel_launch(void* const* d_in, const int* in_sizes, int n_in,
                              void* d_out, int out_size, void* d_ws, size_t ws_size,
                              hipStream_t stream) {
  const float* x      = (const float*)d_in[0];
  const float* c1_ss  = (const float*)d_in[1];
  const float* c1_sv  = (const float*)d_in[2];
  const float* c1_vs  = (const float*)d_in[3];
  const float* c1_vv0 = (const float*)d_in[4];
  const float* c1_vv1 = (const float*)d_in[5];
  const float* c2_ss  = (const float*)d_in[6];
  const float* c2_sv  = (const float*)d_in[7];
  const float* c2_vs  = (const float*)d_in[8];
  const float* c2_vv0 = (const float*)d_in[9];
  const float* c2_vv1 = (const float*)d_in[10];
  const float* w_pos  = (const float*)d_in[11];
  const float* wl_s   = (const float*)d_in[12];
  const float* wl_v   = (const float*)d_in[13];
  float* out = (float*)d_out;

  char* ws = (char*)d_ws;
  bf16*  pb1  = (bf16*)(ws);
  bf16*  pb2  = (bf16*)(ws + PB_BYTES);
  bf16*  xp1  = (bf16*)(ws + 2 * PB_BYTES);
  bf16*  xp2  = (bf16*)(ws + 2 * PB_BYTES + XPAD_BYTES);
  float* ybuf = (float*)(ws + 2 * PB_BYTES + 2 * XPAD_BYTES);
  float* g2   = (float*)(ws);   // reuses pb/xp1 region, dead after conv2

  const int NPB = NTAP * NKC * NNB * 32;            // 72000 threads
  build_pb_kernel<<<(NPB + 255) / 256, 256, 0, stream>>>(
      c1_ss, c1_sv, c1_vs, c1_vv0, c1_vv1, pb1);
  build_pb_kernel<<<(NPB + 255) / 256, 256, 0, stream>>>(
      c2_ss, c2_sv, c2_vs, c2_vv0, c2_vv1, pb2);

  const int NPAD = DP * DP * DP;                    // 46656
  pad_input_kernel<<<(NPAD + 255) / 256, 256, 0, stream>>>(x, xp1);

  dim3 cblk(32, 4);
  const int NBLK = (NVOX / 16) / 4;                 // 512 blocks, 4 waves each
  conv_wmma_kernel<<<NBLK, cblk, 0, stream>>>(xp1, pb1, ybuf);

  gate_pad_kernel<<<(NPAD + 255) / 256, 256, 0, stream>>>(ybuf, xp2);

  conv_wmma_kernel<<<NBLK, cblk, 0, stream>>>(xp2, pb2, ybuf);

  gate_out_kernel<<<(NVOX + 255) / 256, 256, 0, stream>>>(ybuf, g2);

  final_kernel<<<(NFINE + 255) / 256, 256, 0, stream>>>(
      g2, w_pos, wl_s, wl_v, out);
}
